// sl_RGCN_53833120088189
// MI455X (gfx1250) — matrix operand order repaired and checked by reading, not compiled, not run on hardware
//
#include <hip/hip_runtime.h>

#define NN 50000
#define FD 128
#define HD 128
#define RR 8
#define CD 32
#define EE 800000
#define SLOTS 9                 // 8 relations + root(x) as slot 8
#define ROWSTRIDE (SLOTS * FD)  // 1152 bf16 elements per node row
#define NTILES (NN / 16)        // 3125

typedef __attribute__((ext_vector_type(16))) __bf16 v16bf;
typedef __attribute__((ext_vector_type(8)))  float  v8f;

struct __align__(16) U4 { unsigned int x, y, z, w; };

union Frag {
    v16bf v;
    U4 u[2];
    unsigned short s[16];
};

__device__ __forceinline__ unsigned short f2bf(float f) {
    unsigned int u = __float_as_uint(f);
    return (unsigned short)((u + 0x7FFFu + ((u >> 16) & 1u)) >> 16);
}

// ---------------------------------------------------------------------------
// Prep: swizzle weights into WMMA B-fragment layout (bf16), copy x into slot 8
// B-frag layout for 16x16x32 bf16: lane holds col n=lane&15; k = 8*(lane>=16)+16*(j>=8)+(j&7)
// Stored so each lane's 16 values are contiguous: tile*512 + lane*16 + j
// ---------------------------------------------------------------------------
#define WB_ELEMS (SLOTS * 4 * 8 * 512)       // 147456
#define LINWB_ELEMS (4 * 2 * 512)            // 4096
#define XSLOT_ELEMS (NN * FD)                // 6400000
#define PREP_TOTAL (WB_ELEMS + LINWB_ELEMS + XSLOT_ELEMS)

__global__ void prep_kernel(const float* __restrict__ x,
                            const float* __restrict__ W_rel,
                            const float* __restrict__ root,
                            const float* __restrict__ lin_w,
                            unsigned short* __restrict__ tmp9,
                            unsigned short* __restrict__ Wb,
                            unsigned short* __restrict__ linWb) {
    int i = blockIdx.x * blockDim.x + threadIdx.x;
    if (i < WB_ELEMS) {
        int j = i & 15, l = (i >> 4) & 31, t = (i >> 9) & 7, s = (i >> 12) & 3, r = i >> 14;
        int k = s * 32 + ((j >= 8) ? 16 : 0) + ((l >= 16) ? 8 : 0) + (j & 7);
        int col = t * 16 + (l & 15);
        float v = (r < RR) ? W_rel[(r * FD + k) * HD + col] : root[k * HD + col];
        Wb[i] = f2bf(v);
    } else if (i < WB_ELEMS + LINWB_ELEMS) {
        int i2 = i - WB_ELEMS;
        int j = i2 & 15, l = (i2 >> 4) & 31, t = (i2 >> 9) & 1, s = i2 >> 10;
        int k = s * 32 + ((j >= 8) ? 16 : 0) + ((l >= 16) ? 8 : 0) + (j & 7);
        int col = t * 16 + (l & 15);
        linWb[i2] = f2bf(lin_w[k * CD + col]);
    } else if (i < PREP_TOTAL) {
        int i3 = i - (WB_ELEMS + LINWB_ELEMS);
        int n = i3 >> 7, f = i3 & 127;
        tmp9[n * ROWSTRIDE + RR * FD + f] = f2bf(x[i3]);
    }
}

// ---------------------------------------------------------------------------
// Count edges per (dst, relation)
// ---------------------------------------------------------------------------
__global__ void count_kernel(const int* __restrict__ ei,
                             const int* __restrict__ et,
                             unsigned int* __restrict__ counts) {
    int e = blockIdx.x * blockDim.x + threadIdx.x;
    if (e >= EE) return;
    int dst = ei[EE + e];
    int ty  = et[e];
    atomicAdd(&counts[dst * RR + ty], 1u);
}

// ---------------------------------------------------------------------------
// Scatter: tmp9[dst, type, :] += x[src, :] / cnt[dst,type]  (bf16 pk atomics)
// ---------------------------------------------------------------------------
__global__ void scatter_kernel(const float* __restrict__ x,
                               const int* __restrict__ ei,
                               const int* __restrict__ et,
                               const unsigned int* __restrict__ counts,
                               unsigned short* __restrict__ tmp9) {
    int e = blockIdx.x * blockDim.x + threadIdx.x;
    if (e >= EE) return;
    int src = ei[e];
    int dst = ei[EE + e];
    int ty  = et[e];
    unsigned int cnt = counts[dst * RR + ty];
    float norm = 1.0f / (float)(cnt > 0u ? cnt : 1u);
    const float4* xr = (const float4*)(x + (long long)src * FD);
    unsigned long long base =
        (unsigned long long)(tmp9 + (long long)dst * ROWSTRIDE + (long long)ty * FD);
#pragma unroll 4
    for (int q = 0; q < 32; q++) {
        float4 xv = xr[q];
        unsigned int pk0 = (unsigned int)f2bf(xv.x * norm) |
                           ((unsigned int)f2bf(xv.y * norm) << 16);
        unsigned int pk1 = (unsigned int)f2bf(xv.z * norm) |
                           ((unsigned int)f2bf(xv.w * norm) << 16);
        unsigned long long a0 = base + (unsigned long long)(q * 8);
        asm volatile("global_atomic_pk_add_bf16 %0, %1, off scope:SCOPE_DEV"
                     :: "v"(a0), "v"(pk0) : "memory");
        asm volatile("global_atomic_pk_add_bf16 %0, %1, off scope:SCOPE_DEV"
                     :: "v"(a0 + 4ull), "v"(pk1) : "memory");
    }
}

// ---------------------------------------------------------------------------
// Fused GEMM: H = Σ_r tmp9[:,r,:] @ Wb[r]  (+bias, ReLU)  then  out = H @ linW + lin_b
// One wave (wave32) per 16-node tile. All matmuls via v_wmma_f32_16x16x32_bf16.
// __builtin_amdgcn_sched_barrier(0) between the B-load group and the WMMA group
// keeps all 8 B fragments live simultaneously (distinct registers): one wait,
// then 8 back-to-back WMMAs, while the next iteration's load clause is free to
// hoist above the WMMA chain (no barrier on that path).
// ---------------------------------------------------------------------------
__global__ __launch_bounds__(256) void rgcn_gemm_kernel(
    const unsigned short* __restrict__ tmp9,
    const unsigned short* __restrict__ Wb,
    const unsigned short* __restrict__ linWb,
    const float* __restrict__ bias,
    const float* __restrict__ lin_b,
    float* __restrict__ out) {
    __shared__ float hbuf[8][16][HD];  // 64 KB: per-wave transpose staging

    int lane = threadIdx.x & 31;
    int wave = threadIdx.x >> 5;
    int tile = blockIdx.x * 8 + wave;
    if (tile >= NTILES) return;

    int m0   = tile * 16;
    int row  = lane & 15;          // A-frag: row index
    int koff = (lane >> 4) * 8;    // A-frag: k-offset per lane half
    int nn   = lane & 15;          // C/D-frag: column
    int mh   = (lane >> 4) * 8;    // C/D-frag: row half

    v8f acc[8] = {};

    const unsigned short* arow = tmp9 + (long long)(m0 + row) * ROWSTRIDE + koff;
    const U4* wbl = (const U4*)Wb + lane * 2;

#pragma unroll 2
    for (int rs = 0; rs < SLOTS * 4; rs++) {  // rs = r*4 + kstep ; r*128+s*32 == rs*32
        Frag a;
        const unsigned short* ap = arow + rs * 32;
        a.u[0] = *(const U4*)(ap);
        a.u[1] = *(const U4*)(ap + 16);

        const U4* wb = wbl + (long long)rs * 8 * 64;
        Frag b[8];
#pragma unroll
        for (int t = 0; t < 8; t++) {
            b[t].u[0] = wb[t * 64];
            b[t].u[1] = wb[t * 64 + 1];
        }
        // Hard scheduling barrier: nothing crosses. All 18 loads stay above,
        // all 8 WMMAs stay below -> b[0..7] forced simultaneously live.
        __builtin_amdgcn_sched_barrier(0);
#pragma unroll
        for (int t = 0; t < 8; t++) {
            acc[t] = __builtin_amdgcn_wmma_f32_16x16x32_bf16(
                false, a.v, false, b[t].v, (short)0, acc[t], false, false);
        }
    }

    // bias + ReLU, stage into LDS to transpose D-layout -> A-layout
#pragma unroll
    for (int t = 0; t < 8; t++) {
        float bv = bias[t * 16 + nn];
#pragma unroll
        for (int v = 0; v < 8; v++) {
            float val = acc[t][v] + bv;
            hbuf[wave][mh + v][t * 16 + nn] = val > 0.0f ? val : 0.0f;
        }
    }

    // head: [16x128] @ [128x32]
    v8f acc2[2] = {};
#pragma unroll
    for (int s = 0; s < 4; s++) {
        Frag a;
#pragma unroll
        for (int j = 0; j < 8; j++) {
            a.s[j]     = f2bf(hbuf[wave][row][s * 32 + koff + j]);
            a.s[j + 8] = f2bf(hbuf[wave][row][s * 32 + 16 + koff + j]);
        }
#pragma unroll
        for (int c = 0; c < 2; c++) {
            const U4* lp = (const U4*)linWb + (s * 2 + c) * 64 + lane * 2;
            Frag b;
            b.u[0] = lp[0];
            b.u[1] = lp[1];
            acc2[c] = __builtin_amdgcn_wmma_f32_16x16x32_bf16(
                false, a.v, false, b.v, (short)0, acc2[c], false, false);
        }
    }

#pragma unroll
    for (int c = 0; c < 2; c++) {
        float lb = lin_b[c * 16 + nn];
#pragma unroll
        for (int v = 0; v < 8; v++) {
            out[(long long)(m0 + mh + v) * CD + c * 16 + nn] = acc2[c][v] + lb;
        }
    }
}

// ---------------------------------------------------------------------------
// Workspace layout (bytes):
//   counts : [0,          1,600,000)   u32 [N][R]
//   tmp9   : [1,600,000, 116,800,000)  bf16 [N][9][128]
//   Wb     : [116,800,000, 117,094,912) bf16 swizzled weights (9 rel incl root)
//   linWb  : [117,094,912, 117,103,104) bf16 swizzled head weights
// ---------------------------------------------------------------------------
extern "C" void kernel_launch(void* const* d_in, const int* in_sizes, int n_in,
                              void* d_out, int out_size, void* d_ws, size_t ws_size,
                              hipStream_t stream) {
    const float* x     = (const float*)d_in[0];
    const int*   ei    = (const int*)d_in[1];
    const int*   et    = (const int*)d_in[2];
    const float* W_rel = (const float*)d_in[3];
    const float* root  = (const float*)d_in[4];
    const float* bias  = (const float*)d_in[5];
    const float* lin_w = (const float*)d_in[6];
    const float* lin_b = (const float*)d_in[7];
    float* out = (float*)d_out;

    char* ws = (char*)d_ws;
    unsigned int*   counts = (unsigned int*)ws;
    unsigned short* tmp9   = (unsigned short*)(ws + 1600000);
    unsigned short* Wb     = (unsigned short*)(ws + 116800000);
    unsigned short* linWb  = (unsigned short*)(ws + 117094912);

    hipMemsetAsync(ws, 0, 116800000, stream);  // counts + tmp9

    prep_kernel<<<(PREP_TOTAL + 255) / 256, 256, 0, stream>>>(
        x, W_rel, root, lin_w, tmp9, Wb, linWb);

    count_kernel<<<(EE + 255) / 256, 256, 0, stream>>>(ei, et, counts);

    scatter_kernel<<<(EE + 255) / 256, 256, 0, stream>>>(x, ei, et, counts, tmp9);

    rgcn_gemm_kernel<<<(NTILES + 7) / 8, 256, 0, stream>>>(
        tmp9, Wb, linWb, bias, lin_b, out);
}
